// PointConv_76175539962311
// MI455X (gfx1250) — matrix operand order repaired and checked by reading, not compiled, not run on hardware
//
#include <hip/hip_runtime.h>

typedef _Float16 f16;
typedef __attribute__((ext_vector_type(16))) _Float16 v16h;
typedef __attribute__((ext_vector_type(8)))  _Float16 v8h;
typedef __attribute__((ext_vector_type(8)))  float    v8f;

// Problem constants
constexpr int Bn  = 8;
constexpr int Cc  = 64;
constexpr int Nn  = 16384;
constexpr int Kk  = 8;
constexpr int IN1 = 512;   // K*C
constexpr int O1  = 256;
constexpr int O2  = 256;
constexpr int O3  = 128;

// Tiling
constexpr int TN     = 64;        // points per workgroup
constexpr int XPITCH = IN1 + 8;   // LDS row pitch (halves): 520 -> bank spread, 16B aligned rows
constexpr int HPITCH = O1  + 8;   // 264

__device__ __forceinline__ v16h cat16(v8h lo, v8h hi) {
  return __builtin_shufflevector(lo, hi, 0,1,2,3,4,5,6,7,8,9,10,11,12,13,14,15);
}

__device__ __forceinline__ v8f wmma16(v16h a, v16h b, v8f c) {
  return __builtin_amdgcn_wmma_f32_16x16x32_f16(false, a, false, b, (short)0, c,
                                                false, false);
}

// Load the 4 B fragments (one per 16-point column) for K-chunk k0.
__device__ __forceinline__ void load_b4(v16h bm[4], const f16* __restrict__ Xbase,
                                        int pitch, int k0) {
  #pragma unroll
  for (int p = 0; p < 4; p++) {
    const f16* xp = Xbase + p * 16 * pitch + k0;
    bm[p] = cat16(*(const v8h*)(xp), *(const v8h*)(xp + 16));
  }
}

// Two 16x64 output strips (32 rows) in one K-loop, 8 accumulators, with the
// next K-step's B fragments prefetched into registers before this step's wmma.
// Wrow  = W + (orow0 + m) * IN   (strip 0 row; strip 1 row is +16*IN)
// Xbase = X + m * pitch + kb     (lane's activation row base, point column 0)
template<int IN>
__device__ __forceinline__ void gemm_strip2(const f16* __restrict__ Wrow,
                                            const f16* __restrict__ Xbase,
                                            int pitch, int koff, v8f acc[2][4]) {
  v16h bcur[4];
  load_b4(bcur, Xbase, pitch, 0);
  #pragma unroll
  for (int k0 = 0; k0 < IN; k0 += 32) {
    v16h a0 = cat16(*(const v8h*)(Wrow + k0 + koff),
                    *(const v8h*)(Wrow + k0 + koff + 16));
    v16h a1 = cat16(*(const v8h*)(Wrow + 16 * IN + k0 + koff),
                    *(const v8h*)(Wrow + 16 * IN + k0 + koff + 16));
    if (k0 + 32 < IN) {
      v16h bnxt[4];
      load_b4(bnxt, Xbase, pitch, k0 + 32);
      #pragma unroll
      for (int p = 0; p < 4; p++) {
        acc[0][p] = wmma16(a0, bcur[p], acc[0][p]);
        acc[1][p] = wmma16(a1, bcur[p], acc[1][p]);
      }
      #pragma unroll
      for (int p = 0; p < 4; p++) bcur[p] = bnxt[p];   // elided after unroll
    } else {
      #pragma unroll
      for (int p = 0; p < 4; p++) {
        acc[0][p] = wmma16(a0, bcur[p], acc[0][p]);
        acc[1][p] = wmma16(a1, bcur[p], acc[1][p]);
      }
    }
  }
}

// Single 16x64 strip (layer 3): 4 accumulators, same B prefetch.
template<int IN>
__device__ __forceinline__ void gemm_strip(const f16* __restrict__ Wrow,
                                           const f16* __restrict__ Xbase,
                                           int pitch, int koff, v8f acc[4]) {
  v16h bcur[4];
  load_b4(bcur, Xbase, pitch, 0);
  #pragma unroll
  for (int k0 = 0; k0 < IN; k0 += 32) {
    v16h a = cat16(*(const v8h*)(Wrow + k0 + koff),
                   *(const v8h*)(Wrow + k0 + koff + 16));
    if (k0 + 32 < IN) {
      v16h bnxt[4];
      load_b4(bnxt, Xbase, pitch, k0 + 32);
      #pragma unroll
      for (int p = 0; p < 4; p++) acc[p] = wmma16(a, bcur[p], acc[p]);
      #pragma unroll
      for (int p = 0; p < 4; p++) bcur[p] = bnxt[p];
    } else {
      #pragma unroll
      for (int p = 0; p < 4; p++) acc[p] = wmma16(a, bcur[p], acc[p]);
    }
  }
}

// ---------- pre-pass 1: feature [B,C,N] f32 -> featT [B,N,C] f16 (tiled transpose) ----------
__global__ __launch_bounds__(256) void feat_transpose(const float* __restrict__ f,
                                                      f16* __restrict__ ft) {
  __shared__ f16 tile[64 * 66];                 // [c][n], padded
  const int bid = blockIdx.x;                   // Bn * (Nn/64) blocks
  const int b   = bid >> 8;                     // Nn/64 = 256
  const int n0  = (bid & 255) << 6;
  const int tid = threadIdx.x;
  for (int e = tid; e < 64 * 64; e += 256) {    // coalesced read along n
    int c = e >> 6, n = e & 63;
    tile[c * 66 + n] = (f16)f[((size_t)b * Cc + c) * Nn + n0 + n];
  }
  __syncthreads();
  for (int e = tid; e < 64 * 64; e += 256) {    // coalesced write along c
    int n = e >> 6, c = e & 63;
    ft[((size_t)b * Nn + n0 + n) * Cc + c] = tile[c * 66 + n];
  }
}

// ---------- pre-pass 2: weights f32 -> f16 ----------
__global__ __launch_bounds__(256) void cvt_weights(const float* __restrict__ W1,
                                                   const float* __restrict__ W2,
                                                   const float* __restrict__ W3,
                                                   f16* __restrict__ W1h,
                                                   f16* __restrict__ W2h,
                                                   f16* __restrict__ W3h) {
  int t = blockIdx.x * 256 + threadIdx.x;
  if (t < O1 * IN1) W1h[t] = (f16)W1[t];
  if (t < O2 * O1)  W2h[t] = (f16)W2[t];
  if (t < O3 * O2)  W3h[t] = (f16)W3[t];
}

// ---------- main fused kernel: gather + 3 GEMM layers ----------
__global__ __launch_bounds__(256) void pointconv_main(
    const f16* __restrict__ featT, const int* __restrict__ idx32,
    const f16* __restrict__ W1h, const float* __restrict__ b1,
    const f16* __restrict__ W2h, const float* __restrict__ b2,
    const f16* __restrict__ W3h, const float* __restrict__ b3,
    float* __restrict__ out) {
  extern __shared__ __align__(16) char smem[];
  f16* X  = (f16*)smem;                              // [TN][XPITCH]  66,560 B
  f16* H1 = (f16*)(smem + TN * XPITCH * 2);          // [TN][HPITCH]  33,792 B
  f16* H2 = X;                                       // X dead after layer 1

  const int tid   = threadIdx.x;
  const int lane  = tid & 31;
  const int w     = tid >> 5;
  const int bid   = blockIdx.x;                      // Bn * (Nn/TN) = 2048
  const int b     = bid >> 8;                        // Nn/TN = 256
  const int nbase = (bid & 255) * TN;

  // ---- gather: X[p][k*C+c] = featT[b][idx[b][nbase+p][k]][c], 8 halves per load ----
  // idx is int64 but values < 16384: read the low 32-bit word (little-endian).
  for (int e = tid; e < TN * (IN1 / 8); e += 256) {  // 4096 16-byte chunks
    int p  = e >> 6;
    int r8 = e & 63;
    int k  = r8 >> 3;
    int c0 = (r8 & 7) << 3;
    int j  = idx32[(((b * Nn + nbase + p) * Kk) + k) * 2];
    v8h v = *(const v8h*)(featT + ((size_t)b * Nn + (size_t)j) * Cc + c0);
    *(v8h*)(X + p * XPITCH + (r8 << 3)) = v;
  }
  __syncthreads();

  const int m    = lane & 15;            // A row within tile == B column within tile
  const int koff = (lane >> 4) << 3;     // A operand K sub-offset (0 or 8)
  const int kb   = (lane >> 4) << 3;     // B operand K sub-offset (0 or 8)
  const int rhi  = (lane >> 4) << 3;     // D row offset (0 or 8)

  // ---- layer 1: 512 -> 256, bias + relu, to LDS. 32 rows (2 strips) per wave ----
  {
    const int orow0 = w << 5;            // strips 2w, 2w+1
    v8f acc[2][4] = {};
    gemm_strip2<IN1>(W1h + (size_t)(orow0 + m) * IN1, X + m * XPITCH + kb,
                     XPITCH, koff, acc);
    #pragma unroll
    for (int s = 0; s < 2; s++) {
      #pragma unroll
      for (int i = 0; i < 8; i++) {
        const int orow = orow0 + (s << 4) + i + rhi;
        const float bv = b1[orow];
        #pragma unroll
        for (int p = 0; p < 4; p++) {
          float v = fmaxf(acc[s][p][i] + bv, 0.0f);
          H1[(p * 16 + m) * HPITCH + orow] = (f16)v;
        }
      }
    }
  }
  __syncthreads();

  // ---- layer 2: 256 -> 256, bias + relu, to LDS (aliases X). 32 rows per wave ----
  {
    const int orow0 = w << 5;
    v8f acc[2][4] = {};
    gemm_strip2<O1>(W2h + (size_t)(orow0 + m) * O1, H1 + m * HPITCH + kb,
                    HPITCH, koff, acc);
    #pragma unroll
    for (int s = 0; s < 2; s++) {
      #pragma unroll
      for (int i = 0; i < 8; i++) {
        const int orow = orow0 + (s << 4) + i + rhi;
        const float bv = b2[orow];
        #pragma unroll
        for (int p = 0; p < 4; p++) {
          float v = fmaxf(acc[s][p][i] + bv, 0.0f);
          H2[(p * 16 + m) * HPITCH + orow] = (f16)v;
        }
      }
    }
  }
  __syncthreads();

  // ---- layer 3: 256 -> 128, bias, to global [B,128,N] f32. 16 rows per wave ----
  {
    const int orow0 = w << 4;
    v8f acc[4] = {};
    gemm_strip<O2>(W3h + (size_t)(orow0 + m) * O2, H2 + m * HPITCH + kb,
                   HPITCH, koff, acc);
    #pragma unroll
    for (int i = 0; i < 8; i++) {
      const int orow = orow0 + i + rhi;
      const float bv = b3[orow];
      float* orow_ptr = out + ((size_t)b * O3 + orow) * Nn + nbase + m;
      #pragma unroll
      for (int p = 0; p < 4; p++) {
        orow_ptr[p * 16] = acc[p][i] + bv;
      }
    }
  }
}

extern "C" void kernel_launch(void* const* d_in, const int* in_sizes, int n_in,
                              void* d_out, int out_size, void* d_ws, size_t ws_size,
                              hipStream_t stream) {
  const float* feature = (const float*)d_in[0];
  const int*   idx32   = (const int*)d_in[1];     // int64 data, low words used
  const float* W1      = (const float*)d_in[2];
  const float* b1      = (const float*)d_in[3];
  const float* W2      = (const float*)d_in[4];
  const float* b2      = (const float*)d_in[5];
  const float* W3      = (const float*)d_in[6];
  const float* b3      = (const float*)d_in[7];
  float*       out     = (float*)d_out;

  // workspace layout (f16): featT | W1h | W2h | W3h  (~16.5 MB)
  f16* featT = (f16*)d_ws;
  f16* W1h   = featT + (size_t)Bn * Nn * Cc;
  f16* W2h   = W1h + (size_t)O1 * IN1;
  f16* W3h   = W2h + (size_t)O2 * O1;

  feat_transpose<<<Bn * (Nn / 64), 256, 0, stream>>>(feature, featT);
  cvt_weights<<<(O1 * IN1 + 255) / 256, 256, 0, stream>>>(W1, W2, W3, W1h, W2h, W3h);

  size_t smem = (size_t)TN * XPITCH * 2 + (size_t)TN * HPITCH * 2;  // 100,352 B
  hipFuncSetAttribute((const void*)pointconv_main,
                      hipFuncAttributeMaxDynamicSharedMemorySize, (int)smem);
  pointconv_main<<<Bn * (Nn / TN), 256, smem, stream>>>(
      featT, idx32, W1h, b1, W2h, b2, W3h, b3, out);
}
